// DeterministicInformer_6167573037495
// MI455X (gfx1250) — matrix operand order repaired and checked by reading, not compiled, not run on hardware
//
#include <hip/hip_runtime.h>

typedef _Float16 half_t;
typedef __attribute__((ext_vector_type(16))) _Float16 v16h;
typedef __attribute__((ext_vector_type(8)))  _Float16 v8h;
typedef __attribute__((ext_vector_type(8)))  float    v8f;

union FragH { v16h v; v8h h[2]; };
union FragF { v8f  v; float f[8]; };

#define DM    256
#define NH    8
#define DKH   32
#define FFD   1024
#define NB    16
#define L0SEQ 4096
#define NPRED 24

// ---------------------------------------------------------------------------
// WMMA GEMM: C[M,N] = A[M,K](f16) @ W[K,N](f16) + bias, optional exact GELU.
// Block tile 128x128, 8 waves (4Mx2N), wave tile 32x64 (8 x 16x16x32 WMMA).
// A tile staged global->LDS with CDNA5 async loads (ASYNCcnt), B staged
// transposed through VGPRs so fragments are contiguous b128 LDS reads.
// Single barrier per K-step (producer waits own ASYNCcnt/DScnt first).
// ---------------------------------------------------------------------------
#define BM 128
#define BN 128
#define BK 32

__device__ __forceinline__ void async_ld_b128(unsigned lds_off, const void* gptr) {
  asm volatile("global_load_async_to_lds_b128 %0, %1, off"
               :: "v"(lds_off), "v"(gptr) : "memory");
}
__device__ __forceinline__ void wait_async0() {
  asm volatile("s_wait_asynccnt 0x0" ::: "memory");
}

template<int ACT>
__global__ __launch_bounds__(256) void ki_gemm(
    const half_t* __restrict__ A, const half_t* __restrict__ W,
    const float* __restrict__ bias,
    float* __restrict__ Cf, half_t* __restrict__ Ch,
    int M, int N, int K, int lda, int ldb, int ldc,
    long bsA, long bsC)
{
  __shared__ half_t As[2][BM * BK];
  __shared__ half_t Bs[2][BN * BK];

  const int tid  = threadIdx.x;
  const int lane = tid & 31;
  const int wv   = tid >> 5;
  const int wm   = wv & 3;   // 4 waves in M (32 rows each)
  const int wn   = wv >> 2;  // 2 waves in N (64 cols each)
  const int bM   = blockIdx.y * BM;
  const int bN   = blockIdx.x * BN;
  A += (long)blockIdx.z * bsA;
  const long cb0 = (long)blockIdx.z * bsC;

  // staging: A tile 128x32 (16 f16/thread), B tile 32x128 (16 f16/thread)
  const int ar = tid >> 1;
  const int ac = (tid & 1) * 16;
  const int br = tid >> 3;
  const int bc = (tid & 7) * 16;

  v8h b0g, b1g;

  auto asyncA = [&](int k0, int buf) {
    int r = bM + ar;
    unsigned l0 = (unsigned)(unsigned long long)(const void*)&As[buf][ar * BK + ac];
    if (r < M) {
      const half_t* p = A + (long)r * lda + k0 + ac;
      async_ld_b128(l0, p);
      async_ld_b128(l0 + 16, p + 8);
    } else {
      v8h z;
#pragma unroll
      for (int i = 0; i < 8; ++i) z[i] = (_Float16)0.f;
      *(v8h*)&As[buf][ar * BK + ac]     = z;
      *(v8h*)&As[buf][ar * BK + ac + 8] = z;
    }
  };
  auto gloadB = [&](int k0) {
    const half_t* q = W + (long)(k0 + br) * ldb + bN + bc;
    b0g = *(const v8h*)q;
    b1g = *(const v8h*)(q + 8);
  };
  auto stageB = [&](int buf) {
#pragma unroll
    for (int i = 0; i < 8; ++i) Bs[buf][(bc + i) * BK + br] = b0g[i];       // transpose
#pragma unroll
    for (int i = 0; i < 8; ++i) Bs[buf][(bc + 8 + i) * BK + br] = b1g[i];
  };

  FragF acc[2][4];
#pragma unroll
  for (int mi = 0; mi < 2; ++mi)
#pragma unroll
    for (int ni = 0; ni < 4; ++ni)
#pragma unroll
      for (int k = 0; k < 8; ++k) acc[mi][ni].f[k] = 0.f;

  const int nk = K / BK;
  asyncA(0, 0);
  gloadB(0);
  stageB(0);
  wait_async0();
  __syncthreads();

  const int fr = lane & 15;   // M row (A) / N col (B,C) within fragment
  const int kg = lane >> 4;   // lane-half selector

  for (int kt = 0; kt < nk; ++kt) {
    const int cur = kt & 1;
    if (kt + 1 < nk) {
      asyncA((kt + 1) * BK, cur ^ 1);
      gloadB((kt + 1) * BK);
      if (kt + 2 < nk) {  // speculative prefetch of K+2 A tile into caches
        __builtin_prefetch(A + (long)(bM + ar) * lda + (kt + 2) * BK + ac, 0, 1);
      }
    }

    FragH af[2], bf[4];
#pragma unroll
    for (int mi = 0; mi < 2; ++mi) {
      // A layout: lane-half kg holds K = kg*8+0..7 (v0-3), 16+kg*8+0..7 (v4-7)
      const half_t* p = &As[cur][(wm * 32 + mi * 16 + fr) * BK + kg * 8];
      af[mi].h[0] = *(const v8h*)p;
      af[mi].h[1] = *(const v8h*)(p + 16);
    }
#pragma unroll
    for (int ni = 0; ni < 4; ++ni) {
      // B layout: lane-half kg holds K = kg*8+0..7 (v0-3), 16+kg*8+0..7 (v4-7) for column fr
      const half_t* p = &Bs[cur][(wn * 64 + ni * 16 + fr) * BK + kg * 8];
      bf[ni].h[0] = *(const v8h*)p;
      bf[ni].h[1] = *(const v8h*)(p + 16);
    }
#pragma unroll
    for (int mi = 0; mi < 2; ++mi)
#pragma unroll
      for (int ni = 0; ni < 4; ++ni)
        acc[mi][ni].v = __builtin_amdgcn_wmma_f32_16x16x32_f16(
            false, af[mi].v, false, bf[ni].v, (short)0, acc[mi][ni].v, false, false);

    if (kt + 1 < nk) {
      stageB(cur ^ 1);   // buffer cur^1: last read at iteration kt-1, barrier-separated
      wait_async0();     // own async A writes to cur^1 complete before signaling
    }
    __syncthreads();
  }

  // epilogue: C frag -> row = v + kg*8, col = fr
#pragma unroll
  for (int mi = 0; mi < 2; ++mi) {
#pragma unroll
    for (int ni = 0; ni < 4; ++ni) {
      int col = bN + wn * 64 + ni * 16 + fr;
      if (col >= N) continue;
      float bv = bias ? bias[col] : 0.f;
#pragma unroll
      for (int v = 0; v < 8; ++v) {
        int row = bM + wm * 32 + mi * 16 + kg * 8 + v;
        if (row >= M) continue;
        float val = acc[mi][ni].f[v] + bv;
        if (ACT == 1) val = 0.5f * val * (1.f + erff(val * 0.70710678118654752f));
        long o = cb0 + (long)row * ldc + col;
        if (Cf) Cf[o] = val;
        if (Ch) Ch[o] = (half_t)val;
      }
    }
  }
}

// ---------------------------------------------------------------------------
// Small kernels
// ---------------------------------------------------------------------------
__global__ void ki_cvt(const float* __restrict__ in, half_t* __restrict__ out, long n) {
  long i = (long)blockIdx.x * 256 + threadIdx.x;
  if (i < n) out[i] = (half_t)in[i];
}

// cw [2][O=256][I=256][T=3] -> out [2][t*256+i][o] (f16)
__global__ void ki_convw(const float* __restrict__ cw, half_t* __restrict__ out, int n) {
  int i = blockIdx.x * 256 + threadIdx.x;
  if (i >= n) return;
  int o = i & 255;
  int t2 = i >> 8;
  int kk = t2 % 768;
  int layer = t2 / 768;
  int t = kk >> 8;
  int ci = kk & 255;
  out[i] = (half_t)cw[(((long)layer * 256 + o) * 256 + ci) * 3 + t];
}

__global__ void ki_embed(const float* __restrict__ x, const float* __restrict__ w,
                         const float* __restrict__ b, const float* __restrict__ pos,
                         float* __restrict__ h, half_t* __restrict__ h16, int L)
{
  long i = (long)blockIdx.x * 256 + threadIdx.x;
  int c = (int)(i & 255);
  long bl = i >> 8;
  int l = (int)(bl % L);
  const float* xr = x + bl * 8;
  float s = b[c] + pos[(long)l * 256 + c];
#pragma unroll
  for (int k = 0; k < 8; ++k) s += xr[k] * w[k * 256 + c];
  h[i] = s; h16[i] = (half_t)s;
}

__global__ void ki_gather(const half_t* __restrict__ K16, const int* __restrict__ idx,
                          half_t* __restrict__ Ks, int L, int u)
{
  int i = blockIdx.x * 256 + threadIdx.x;
  int n = NB * NH * u * DKH;
  if (i >= n) return;
  int d = i & 31;
  int j = (i >> 5) % u;
  int bh = i / (32 * u);
  int b = bh >> 3, h = bh & 7;
  int t = idx[j];
  Ks[i] = K16[((long)(b * L + t)) * 256 + h * 32 + d];
}

// M[b,h,l] = max_j(score) - mean_j(score), score = scale * Q . Ks_j
__global__ void ki_probe(const half_t* __restrict__ Q16, const half_t* __restrict__ Ks,
                         float* __restrict__ Mb, int L, int u, float scale)
{
  long i = (long)blockIdx.x * 256 + threadIdx.x;
  long n = (long)NB * NH * L;
  if (i >= n) return;
  int l = (int)(i % L);
  int bh = (int)(i / L);
  int b = bh >> 3, h = bh & 7;
  const half_t* qp = Q16 + ((long)(b * L + l)) * 256 + h * 32;
  float q[32];
#pragma unroll
  for (int d = 0; d < 32; ++d) q[d] = (float)qp[d];
  const half_t* ks = Ks + (long)bh * u * 32;
  float mx = -3.4e38f, sm = 0.f;
  for (int j = 0; j < u; ++j) {
    float s = 0.f;
#pragma unroll
    for (int d = 0; d < 32; ++d) s += q[d] * (float)ks[j * 32 + d];
    s *= scale;
    mx = fmaxf(mx, s);
    sm += s;
  }
  Mb[i] = mx - sm / (float)u;
}

// iterative top-u (descending, tie -> lowest index); destroys Mb (scratch)
__global__ void ki_topk(float* __restrict__ Mb, int* __restrict__ top, int L, int u)
{
  __shared__ float sv[256];
  __shared__ int   si[256];
  const int tid = threadIdx.x;
  const int bh = blockIdx.x;
  float* Mr = Mb + (long)bh * L;
  for (int j = 0; j < u; ++j) {
    float best = -3.4e38f; int bi = 0x7fffffff;
    for (int l = tid; l < L; l += 256) {
      float v = Mr[l];
      if (v > best) { best = v; bi = l; }
    }
    sv[tid] = best; si[tid] = bi;
    __syncthreads();
    for (int s = 128; s > 0; s >>= 1) {
      if (tid < s) {
        if (sv[tid + s] > sv[tid] || (sv[tid + s] == sv[tid] && si[tid + s] < si[tid])) {
          sv[tid] = sv[tid + s]; si[tid] = si[tid + s];
        }
      }
      __syncthreads();
    }
    if (tid == 0) { top[bh * u + j] = si[0]; Mr[si[0]] = -3.4e38f; }
    __syncthreads();
  }
}

// one block per (b,h,j): softmax(Qt.K) @ V streamed over L
__global__ __launch_bounds__(256) void ki_attn(
    const half_t* __restrict__ Q16, const half_t* __restrict__ K16,
    const half_t* __restrict__ V16, const int* __restrict__ top,
    float* __restrict__ ctxtop, int L, int u, float scale)
{
  __shared__ float qt[32];
  __shared__ float red[8];
  __shared__ float cp[8 * 32];
  const int tid = threadIdx.x;
  const int lane = tid & 31, wv = tid >> 5;
  const int j = blockIdx.x % u;
  const int bh = blockIdx.x / u;
  const int b = bh >> 3, h = bh & 7;
  const int t = top[bh * u + j];
  if (tid < 32) qt[tid] = (float)Q16[((long)(b * L + t)) * 256 + h * 32 + tid];
  __syncthreads();
  const int iters = L >> 8;
  float sc[16];
  float mx = -3.4e38f;
  for (int it = 0; it < iters; ++it) {
    int l = (it << 8) + tid;
    const half_t* kp = K16 + ((long)(b * L + l)) * 256 + h * 32;
    float s = 0.f;
#pragma unroll
    for (int d = 0; d < 32; ++d) s += qt[d] * (float)kp[d];
    s *= scale;
    sc[it] = s;
    mx = fmaxf(mx, s);
  }
#pragma unroll
  for (int o = 16; o > 0; o >>= 1) mx = fmaxf(mx, __shfl_down(mx, o, 32));
  if (lane == 0) red[wv] = mx;
  __syncthreads();
  if (tid == 0) { float m = red[0]; for (int w = 1; w < 8; ++w) m = fmaxf(m, red[w]); red[0] = m; }
  __syncthreads();
  mx = red[0];
  __syncthreads();
  float ssum = 0.f;
  float ca[32];
#pragma unroll
  for (int d = 0; d < 32; ++d) ca[d] = 0.f;
  for (int it = 0; it < iters; ++it) {
    int l = (it << 8) + tid;
    float e = expf(sc[it] - mx);
    ssum += e;
    const half_t* vp = V16 + ((long)(b * L + l)) * 256 + h * 32;
#pragma unroll
    for (int d = 0; d < 32; ++d) ca[d] += e * (float)vp[d];
  }
#pragma unroll
  for (int o = 16; o > 0; o >>= 1) ssum += __shfl_down(ssum, o, 32);
  if (lane == 0) red[wv] = ssum;
  __syncthreads();
  if (tid == 0) { float s = 0.f; for (int w = 0; w < 8; ++w) s += red[w]; red[0] = s; }
  __syncthreads();
  ssum = red[0];
#pragma unroll
  for (int d = 0; d < 32; ++d) {
    float v = ca[d];
#pragma unroll
    for (int o = 16; o > 0; o >>= 1) v += __shfl_down(v, o, 32);
    if (lane == 0) cp[wv * 32 + d] = v;
  }
  __syncthreads();
  if (tid < 32) {
    float v = 0.f;
    for (int w = 0; w < 8; ++w) v += cp[w * 32 + tid];
    ctxtop[((long)bh * u + j) * 32 + tid] = v / ssum;
  }
}

__global__ void ki_vmean(const half_t* __restrict__ V16, float* __restrict__ vm, int L) {
  const int b = blockIdx.x, c = threadIdx.x;
  float s = 0.f;
  for (int l = 0; l < L; ++l) s += (float)V16[((long)(b * L + l)) * 256 + c];
  vm[b * 256 + c] = s / (float)L;
}

__global__ void ki_ctxinit(const float* __restrict__ vm, half_t* __restrict__ ctx, int L) {
  long i = (long)blockIdx.x * 256 + threadIdx.x;
  int c = (int)(i & 255);
  int b = (int)((i >> 8) / L);
  ctx[i] = (half_t)vm[b * 256 + c];
}

// sequential over j -> deterministic last-write-wins (matches scatter set order)
__global__ void ki_scatter(const int* __restrict__ top, const float* __restrict__ ctxtop,
                           half_t* __restrict__ ctx, int L, int u) {
  const int b = blockIdx.x, c = threadIdx.x;
  const int h = c >> 5, d = c & 31;
  for (int j = 0; j < u; ++j) {
    int t = top[(b * 8 + h) * u + j];
    ctx[((long)(b * L + t)) * 256 + c] =
        (half_t)ctxtop[((long)(b * 8 + h) * u + j) * 32 + d];
  }
}

__global__ __launch_bounds__(256) void ki_addln(
    float* __restrict__ h, const float* __restrict__ a,
    const float* __restrict__ g, const float* __restrict__ bb,
    half_t* __restrict__ h16) {
  __shared__ float red[8];
  const long row = blockIdx.x;
  const int c = threadIdx.x;
  const int lane = c & 31, wv = c >> 5;
  long i = row * 256 + c;
  float x = h[i] + a[i];
  float s = x;
#pragma unroll
  for (int o = 16; o > 0; o >>= 1) s += __shfl_down(s, o, 32);
  if (lane == 0) red[wv] = s;
  __syncthreads();
  if (c == 0) { float t = 0.f; for (int w = 0; w < 8; ++w) t += red[w]; red[0] = t; }
  __syncthreads();
  float mean = red[0] * (1.f / 256.f);
  __syncthreads();
  float dv = x - mean;
  float q = dv * dv;
#pragma unroll
  for (int o = 16; o > 0; o >>= 1) q += __shfl_down(q, o, 32);
  if (lane == 0) red[wv] = q;
  __syncthreads();
  if (c == 0) { float t = 0.f; for (int w = 0; w < 8; ++w) t += red[w]; red[0] = t; }
  __syncthreads();
  float var = red[0] * (1.f / 256.f);
  float y = dv * rsqrtf(var + 1e-5f) * g[c] + bb[c];
  h[i] = y;
  h16[i] = (half_t)y;
}

__global__ void ki_pad(const half_t* __restrict__ h16, half_t* __restrict__ pad, int L) {
  const int c = threadIdx.x;
  const long rb = blockIdx.x;              // b*(L+2)+r
  const int r = (int)(rb % (L + 2));
  const int b = (int)(rb / (L + 2));
  half_t v = (half_t)0.f;
  if (r > 0 && r <= L) v = h16[((long)(b * L + r - 1)) * 256 + c];
  pad[rb * 256 + c] = v;
}

__global__ void ki_elupool(const float* __restrict__ y, float* __restrict__ h,
                           half_t* __restrict__ h16, int L) {
  const long i = (long)blockIdx.x * 256 + threadIdx.x;
  const int c = (int)(i & 255);
  const long t = i >> 8;
  const int Lo = L >> 1;
  const int lo = (int)(t % Lo);
  const int b = (int)(t / Lo);
  const float* yp = y + ((long)(b * L + 2 * lo)) * 256 + c;
  float a0 = yp[0];   a0 = a0 > 0.f ? a0 : expm1f(a0);
  float a1 = yp[256]; a1 = a1 > 0.f ? a1 : expm1f(a1);
  float m = fmaxf(a0, a1);
  h[i] = m;
  h16[i] = (half_t)m;
}

__global__ void ki_final(const float* __restrict__ h, const float* __restrict__ fcw,
                         const float* __restrict__ fcb, float* __restrict__ out, int Lf) {
  int i = blockIdx.x * 256 + threadIdx.x;
  if (i >= NB * NPRED) return;
  int b = i / NPRED, p = i % NPRED;
  const float* hr = h + ((long)(b * Lf + Lf - 1)) * 256;
  float s = fcb[p];
  for (int c = 0; c < 256; ++c) s += hr[c] * fcw[c * NPRED + p];
  out[i] = s;
}

// ---------------------------------------------------------------------------
static void gemm(hipStream_t s, const half_t* A, const half_t* W, const float* bias,
                 float* Cf, half_t* Ch, int M, int N, int K,
                 int lda, int ldb, int ldc, long bsA, long bsC, int batches, int act)
{
  dim3 g((N + BN - 1) / BN, (M + BM - 1) / BM, batches);
  if (act)
    ki_gemm<1><<<g, 256, 0, s>>>(A, W, bias, Cf, Ch, M, N, K, lda, ldb, ldc, bsA, bsC);
  else
    ki_gemm<0><<<g, 256, 0, s>>>(A, W, bias, Cf, Ch, M, N, K, lda, ldb, ldc, bsA, bsC);
}

extern "C" void kernel_launch(void* const* d_in, const int* in_sizes, int n_in,
                              void* d_out, int out_size, void* d_ws, size_t ws_size,
                              hipStream_t stream)
{
  (void)n_in; (void)out_size; (void)ws_size;
  const float* x   = (const float*)d_in[0];
  const int* idx_[3] = {(const int*)d_in[1], (const int*)d_in[2], (const int*)d_in[3]};
  const int  u_[3]   = { in_sizes[1], in_sizes[2], in_sizes[3] };
  const float* in_w = (const float*)d_in[4];
  const float* in_b = (const float*)d_in[5];
  const float* pos  = (const float*)d_in[6];
  const float* qw = (const float*)d_in[7];  const float* qb = (const float*)d_in[8];
  const float* kw = (const float*)d_in[9];  const float* kb = (const float*)d_in[10];
  const float* vw = (const float*)d_in[11]; const float* vb = (const float*)d_in[12];
  const float* ow = (const float*)d_in[13]; const float* ob = (const float*)d_in[14];
  const float* f1w = (const float*)d_in[15]; const float* f1b = (const float*)d_in[16];
  const float* f2w = (const float*)d_in[17]; const float* f2b = (const float*)d_in[18];
  const float* n1g = (const float*)d_in[19]; const float* n1b = (const float*)d_in[20];
  const float* n2g = (const float*)d_in[21]; const float* n2b = (const float*)d_in[22];
  const float* cw  = (const float*)d_in[23]; const float* cb  = (const float*)d_in[24];
  const float* fcw = (const float*)d_in[25]; const float* fcb = (const float*)d_in[26];
  float* out = (float*)d_out;

  // bump allocator over d_ws
  char* wsp = (char*)d_ws;
  size_t off = 0;
  auto alloc = [&](size_t bytes) -> void* {
    void* p = wsp + off;
    off += (bytes + 255) & ~(size_t)255;
    return p;
  };
  float*  hF   = (float*)alloc((size_t)NB * L0SEQ * DM * 4);
  float*  tF   = (float*)alloc((size_t)NB * L0SEQ * DM * 4);
  half_t* hH   = (half_t*)alloc((size_t)NB * L0SEQ * DM * 2);
  half_t* midH = (half_t*)alloc((size_t)NB * L0SEQ * FFD * 2);
  half_t* qH   = (half_t*)alloc((size_t)NB * L0SEQ * DM * 2);
  half_t* kH   = (half_t*)alloc((size_t)NB * L0SEQ * DM * 2);
  half_t* vH   = (half_t*)alloc((size_t)NB * L0SEQ * DM * 2);
  half_t* ctxH = (half_t*)alloc((size_t)NB * L0SEQ * DM * 2);
  half_t* padH = (half_t*)alloc((size_t)NB * (L0SEQ + 2) * DM * 2);
  half_t* ksH  = (half_t*)alloc((size_t)NB * NH * 64 * DKH * 2);
  float*  Mf   = (float*)alloc((size_t)NB * NH * L0SEQ * 4);
  int*    topI = (int*)alloc((size_t)NB * NH * 64 * 4);
  float*  ctF  = (float*)alloc((size_t)NB * NH * 64 * DKH * 4);
  float*  vmF  = (float*)alloc((size_t)NB * DM * 4);
  half_t* qw16 = (half_t*)alloc((size_t)3 * DM * DM * 2);
  half_t* kw16 = (half_t*)alloc((size_t)3 * DM * DM * 2);
  half_t* vw16 = (half_t*)alloc((size_t)3 * DM * DM * 2);
  half_t* ow16 = (half_t*)alloc((size_t)3 * DM * DM * 2);
  half_t* f1w16 = (half_t*)alloc((size_t)3 * DM * FFD * 2);
  half_t* f2w16 = (half_t*)alloc((size_t)3 * FFD * DM * 2);
  half_t* cw16 = (half_t*)alloc((size_t)2 * 3 * DM * DM * 2);

  // weight conversions (deterministic, redone each call)
  {
    long n1 = 3L * DM * DM;
    ki_cvt<<<(int)((n1 + 255) / 256), 256, 0, stream>>>(qw, qw16, n1);
    ki_cvt<<<(int)((n1 + 255) / 256), 256, 0, stream>>>(kw, kw16, n1);
    ki_cvt<<<(int)((n1 + 255) / 256), 256, 0, stream>>>(vw, vw16, n1);
    ki_cvt<<<(int)((n1 + 255) / 256), 256, 0, stream>>>(ow, ow16, n1);
    long n2 = 3L * DM * FFD;
    ki_cvt<<<(int)((n2 + 255) / 256), 256, 0, stream>>>(f1w, f1w16, n2);
    ki_cvt<<<(int)((n2 + 255) / 256), 256, 0, stream>>>(f2w, f2w16, n2);
    int n3 = 2 * 3 * DM * DM;
    ki_convw<<<(n3 + 255) / 256, 256, 0, stream>>>(cw, cw16, n3);
  }

  // input embedding + positional
  {
    long n = (long)NB * L0SEQ * DM;
    ki_embed<<<(int)(n / 256), 256, 0, stream>>>(x, in_w, in_b, pos, hF, hH, L0SEQ);
  }

  const float scale = 0.17677669529663687f; // 1/sqrt(32)
  int L = L0SEQ;
  for (int layer = 0; layer < 3; ++layer) {
    const int M = NB * L;
    const int u = u_[layer];
    const int* idx = idx_[layer];
    const long woff = (long)layer * DM * DM;

    // Q,K,V projections (f16 out)
    gemm(stream, hH, qw16 + woff, qb + layer * DM, nullptr, qH, M, DM, DM, DM, DM, DM, 0, 0, 1, 0);
    gemm(stream, hH, kw16 + woff, kb + layer * DM, nullptr, kH, M, DM, DM, DM, DM, DM, 0, 0, 1, 0);
    gemm(stream, hH, vw16 + woff, vb + layer * DM, nullptr, vH, M, DM, DM, DM, DM, DM, 0, 0, 1, 0);

    // ProbSparse selection
    {
      int ng = NB * NH * u * DKH;
      ki_gather<<<(ng + 255) / 256, 256, 0, stream>>>(kH, idx, ksH, L, u);
      long np = (long)NB * NH * L;
      ki_probe<<<(int)((np + 255) / 256), 256, 0, stream>>>(qH, ksH, Mf, L, u, scale);
      ki_topk<<<NB * NH, 256, 0, stream>>>(Mf, topI, L, u);
      ki_attn<<<NB * NH * u, 256, 0, stream>>>(qH, kH, vH, topI, ctF, L, u, scale);
      ki_vmean<<<NB, 256, 0, stream>>>(vH, vmF, L);
      ki_ctxinit<<<M, 256, 0, stream>>>(vmF, ctxH, L);
      ki_scatter<<<NB, 256, 0, stream>>>(topI, ctF, ctxH, L, u);
    }

    // output projection + residual LN
    gemm(stream, ctxH, ow16 + woff, ob + layer * DM, tF, nullptr, M, DM, DM, DM, DM, DM, 0, 0, 1, 0);
    ki_addln<<<M, 256, 0, stream>>>(hF, tF, n1g + layer * DM, n1b + layer * DM, hH);

    // FFN: gelu(h @ f1w + b) @ f2w + b, residual LN
    gemm(stream, hH, f1w16 + (long)layer * DM * FFD, f1b + layer * FFD, nullptr, midH,
         M, FFD, DM, DM, FFD, FFD, 0, 0, 1, 1);
    gemm(stream, midH, f2w16 + (long)layer * FFD * DM, f2b + layer * DM, tF, nullptr,
         M, DM, FFD, FFD, DM, DM, 0, 0, 1, 0);
    ki_addln<<<M, 256, 0, stream>>>(hF, tF, n2g + layer * DM, n2b + layer * DM, hH);

    // distill: conv1d(k=3,pad=1) as one K=768 GEMM over padded rows, ELU, maxpool/2
    if (layer < 2) {
      ki_pad<<<NB * (L + 2), 256, 0, stream>>>(hH, padH, L);
      gemm(stream, padH, cw16 + (long)layer * 3 * DM * DM, cb + layer * DM, tF, nullptr,
           L, DM, 3 * DM, DM, DM, DM, (long)(L + 2) * DM, (long)L * DM, NB, 0);
      ki_elupool<<<NB * (L / 2), 256, 0, stream>>>(tF, hF, hH, L);
      L >>= 1;
    }
  }

  // head: h[:, -1, :] @ fcw + fcb
  ki_final<<<2, 256, 0, stream>>>(hF, fcw, fcb, out, L);
}
